// SparseFlashAttn_69234872812253
// MI455X (gfx1250) — matrix operand order repaired and checked by reading, not compiled, not run on hardware
//
#include <hip/hip_runtime.h>

#define BATCH 8
#define HEADS 32
#define HKV 8
#define GQ 4
#define DIM 128
#define DIMV 128
#define PAGE_BLK 256
#define BLOCK_N 64
#define MAX_SEL 64
#define PAGES_PER_SEQ 16
#define NWAVES 8
#define NTHREADS (NWAVES * 32)

typedef __attribute__((ext_vector_type(16))) __bf16 v16bf;
typedef __attribute__((ext_vector_type(8)))  __bf16 v8bf;
typedef __attribute__((ext_vector_type(16))) float  v16f;
typedef __attribute__((ext_vector_type(8)))  float  v8f;
typedef __attribute__((ext_vector_type(4)))  float  v4f;

// 16 contiguous fp32 -> v16bf: 4 NT b128 loads + one 16-wide fptrunc
// (legalizes to 8 v_cvt_pk_bf16_f32, pure dataflow -> scheduler keeps MLP)
__device__ __forceinline__ v16bf cvt16c(const float* p) {
  v4f x0 = __builtin_nontemporal_load((const v4f*)p + 0);
  v4f x1 = __builtin_nontemporal_load((const v4f*)p + 1);
  v4f x2 = __builtin_nontemporal_load((const v4f*)p + 2);
  v4f x3 = __builtin_nontemporal_load((const v4f*)p + 3);
  v8f y0 = __builtin_shufflevector(x0, x1, 0, 1, 2, 3, 4, 5, 6, 7);
  v8f y1 = __builtin_shufflevector(x2, x3, 0, 1, 2, 3, 4, 5, 6, 7);
  v16f z = __builtin_shufflevector(y0, y1,
           0, 1, 2, 3, 4, 5, 6, 7, 8, 9, 10, 11, 12, 13, 14, 15);
  return __builtin_convertvector(z, v16bf);
}

// 16 strided fp32 -> v16bf: 16 NT dword loads (one clause) + 16-wide fptrunc
__device__ __forceinline__ v16bf cvt16s(const float* p, size_t stride) {
  v16f z;
#pragma unroll
  for (int j = 0; j < 16; ++j)
    z[j] = __builtin_nontemporal_load(p + (size_t)j * stride);
  return __builtin_convertvector(z, v16bf);
}

// reduce across each 16-lane half of the wave32 (masks 1,2,4,8)
__device__ __forceinline__ float halfmax(float v) {
#pragma unroll
  for (int m = 1; m <= 8; m <<= 1) v = fmaxf(v, __shfl_xor(v, m, 32));
  return v;
}
__device__ __forceinline__ float halfsum(float v) {
#pragma unroll
  for (int m = 1; m <= 8; m <<= 1) v += __shfl_xor(v, m, 32);
  return v;
}

__global__ __launch_bounds__(NTHREADS, 1)
void sparse_flash_attn_kernel(const float* __restrict__ query,
                              const float* __restrict__ kcache,
                              const float* __restrict__ vcache,
                              const int*   __restrict__ bidx,
                              const int*   __restrict__ seqlens,
                              const int*   __restrict__ btab,
                              float*       __restrict__ out) {
  const int wg   = blockIdx.x;
  const int b    = wg / HKV;
  const int h    = wg % HKV;
  const int tid  = threadIdx.x;
  const int wave = tid >> 5;
  const int lane = tid & 31;
  const int n    = lane & 15;          // column / row-within-half index
  const bool lohalf = lane < 16;

  __shared__ __attribute__((aligned(16))) __bf16 sP[NWAVES][16 * BLOCK_N];
  __shared__ float sO[NWAVES][GQ][DIMV];
  __shared__ float sM[NWAVES][GQ];
  __shared__ float sL[NWAVES][GQ];

  // zero the P staging buffers (rows 4..15 must stay zero forever)
  for (int i = tid; i < NWAVES * 16 * BLOCK_N; i += NTHREADS)
    (&sP[0][0])[i] = __builtin_bit_cast(__bf16, (unsigned short)0);
  __syncthreads();

  const int   seqlen    = seqlens[b];
  const float inv_scale = 0.08838834764831845f;   // 1/sqrt(128)

  const int r    = n;                  // A-fragment row this lane owns
  const int aoff = lohalf ? 0 : 16;    // A-layout K offset
  const int koff = lohalf ? 0 : 16;    // B-layout K offset

  const v8f zero8 = {0.f, 0.f, 0.f, 0.f, 0.f, 0.f, 0.f, 0.f};

  // ---- Q fragments (A layout, bf16); rows >= GQ zeroed ----
  v16bf Aq[4];
  {
    const float* qp = query + (size_t)(b * HEADS + h * GQ + r) * DIM;
#pragma unroll
    for (int kk = 0; kk < 4; ++kk) {
      const int k0 = kk * 32;
      v16f z;
#pragma unroll
      for (int j = 0; j < 8; ++j) {
        z[j]     = (r < GQ) ? qp[k0 + aoff + j]      : 0.0f;
        z[8 + j] = (r < GQ) ? qp[k0 + 8 + aoff + j] : 0.0f;
      }
      Aq[kk] = __builtin_convertvector(z, v16bf);
    }
  }

  v8f Oacc[8];
#pragma unroll
  for (int i = 0; i < 8; ++i) Oacc[i] = zero8;

  float m_run[GQ], l_run[GQ];
#pragma unroll
  for (int q = 0; q < GQ; ++q) { m_run[q] = -__builtin_inff(); l_run[q] = 0.0f; }

  __bf16* sPw = &sP[wave][0];
  const size_t krow = (size_t)HKV * DIM;    // 1024 floats per token row
  const size_t vrow = (size_t)HKV * DIMV;
  const size_t ibase = ((size_t)b * HKV + h) * MAX_SEL;

  for (int sel = wave; sel < MAX_SEL; sel += NWAVES) {
    const int blk = bidx[ibase + sel];
    if (blk < 0) break;                    // tail is all -1: per-wave break is safe
    const int tok0 = blk * BLOCK_N;
    const int page = btab[b * PAGES_PER_SEQ + (tok0 / PAGE_BLK)];
    const int poff = tok0 % PAGE_BLK;      // 64-token block never crosses a page
    const float* kbase = kcache + (((size_t)page * PAGE_BLK + poff) * HKV + h) * DIM;
    const float* vbase = vcache + (((size_t)page * PAGE_BLK + poff) * HKV + h) * DIMV;

    // ---- warm L2/TLB for the NEXT selected block (global_prefetch_b8) ----
    if (sel + NWAVES < MAX_SEL) {
      const int nblk = bidx[ibase + sel + NWAVES];
      if (nblk >= 0) {
        const int ntok0 = nblk * BLOCK_N;
        const int npage = btab[b * PAGES_PER_SEQ + (ntok0 / PAGE_BLK)];
        const int npoff = ntok0 % PAGE_BLK;
        const float* nk = kcache + (((size_t)npage * PAGE_BLK + npoff) * HKV + h) * DIM;
        const float* nv = vcache + (((size_t)npage * PAGE_BLK + npoff) * HKV + h) * DIMV;
        __builtin_prefetch(nk + (size_t)(lane * 2)     * krow, 0, 0);
        __builtin_prefetch(nk + (size_t)(lane * 2 + 1) * krow, 0, 0);
        __builtin_prefetch(nv + (size_t)(lane * 2)     * vrow, 0, 0);
        __builtin_prefetch(nv + (size_t)(lane * 2 + 1) * vrow, 0, 0);
      }
    }

    // ---- S = Q * K^T : four 16-token tiles, K=128 in 4 WMMA steps ----
    v8f S[4];
#pragma unroll
    for (int tt = 0; tt < 4; ++tt) {
      S[tt] = zero8;
      const float* kt = kbase + (size_t)(tt * 16 + n) * krow + koff;
#pragma unroll
      for (int kk = 0; kk < 4; ++kk) {
        v16bf Bk = cvt16c(kt + kk * 32);
        S[tt] = __builtin_amdgcn_wmma_f32_16x16x32_bf16(
            false, Aq[kk], false, Bk, (short)0, S[tt], false, false);
      }
    }

    // ---- logits + length mask ----
#pragma unroll
    for (int tt = 0; tt < 4; ++tt) {
      const bool valid = (tok0 + tt * 16 + n) < seqlen;
#pragma unroll
      for (int q = 0; q < GQ; ++q)
        S[tt][q] = valid ? S[tt][q] * inv_scale : -__builtin_inff();
    }

    // ---- online softmax row stats; P -> LDS (A-layout staging) ----
    float alpha[GQ];
#pragma unroll
    for (int q = 0; q < GQ; ++q) {
      float mx = S[0][q];
#pragma unroll
      for (int tt = 1; tt < 4; ++tt) mx = fmaxf(mx, S[tt][q]);
      mx = halfmax(mx);
      const float mnew = fmaxf(m_run[q], mx);
      const float a    = __expf(m_run[q] - mnew);   // exp(-inf)=0 on first block
      float ps = 0.0f;
#pragma unroll
      for (int tt = 0; tt < 4; ++tt) {
        const float p = __expf(S[tt][q] - mnew);
        ps += p;
        if (lohalf) sPw[q * BLOCK_N + tt * 16 + n] = (__bf16)p;
      }
      ps = halfsum(ps);
      l_run[q] = l_run[q] * a + ps;
      m_run[q] = mnew;
      alpha[q] = a;
    }

    // CDNA5 split-counter wait: make P stores visible to this wave's ds loads
    asm volatile("s_wait_dscnt 0" ::: "memory");

    // ---- reload P as A fragments (16B LDS chunks) ----
    v16bf Ap[2];
#pragma unroll
    for (int kk = 0; kk < 2; ++kk) {
      const int k0 = kk * 32;
      v8bf lo = *(const v8bf*)&sPw[r * BLOCK_N + k0 + aoff];
      v8bf hi = *(const v8bf*)&sPw[r * BLOCK_N + k0 + 8 + aoff];
      Ap[kk] = __builtin_shufflevector(lo, hi,
               0, 1, 2, 3, 4, 5, 6, 7, 8, 9, 10, 11, 12, 13, 14, 15);
    }

    // ---- rescale accumulators (only valid rows 0..3 matter; rest are 0) ----
#pragma unroll
    for (int nv = 0; nv < 8; ++nv)
#pragma unroll
      for (int v = 0; v < 4; ++v)
        Oacc[nv][v] *= alpha[v];

    // ---- O += P * V : K=64 in 2 WMMA steps, 8 dv tiles ----
#pragma unroll
    for (int nv = 0; nv < 8; ++nv) {
#pragma unroll
      for (int kk = 0; kk < 2; ++kk) {
        const float* vp = vbase + (size_t)(kk * 32 + koff) * vrow + nv * 16 + n;
        v16bf Bv = cvt16s(vp, vrow);
        Oacc[nv] = __builtin_amdgcn_wmma_f32_16x16x32_bf16(
            false, Ap[kk], false, Bv, (short)0, Oacc[nv], false, false);
      }
    }
  }

  // ---- publish per-wave partials ----
  if (lohalf) {
#pragma unroll
    for (int q = 0; q < GQ; ++q) {
#pragma unroll
      for (int nv = 0; nv < 8; ++nv)
        sO[wave][q][nv * 16 + n] = Oacc[nv][q];
      if (n == 0) { sM[wave][q] = m_run[q]; sL[wave][q] = l_run[q]; }
    }
  }
  __syncthreads();

  // ---- merge the wave partials (flash combine) and write out ----
  for (int idx = tid; idx < GQ * DIMV; idx += NTHREADS) {
    const int q  = idx >> 7;
    const int dv = idx & (DIMV - 1);
    float M = sM[0][q];
#pragma unroll
    for (int w = 1; w < NWAVES; ++w) M = fmaxf(M, sM[w][q]);
    float L = 0.0f, acc = 0.0f;
#pragma unroll
    for (int w = 0; w < NWAVES; ++w) {
      const float wgt = __expf(sM[w][q] - M);
      L   += wgt * sL[w][q];
      acc += wgt * sO[w][q][dv];
    }
    out[(size_t)(b * HEADS + h * GQ + q) * DIMV + dv] = acc / L;
  }
}

extern "C" void kernel_launch(void* const* d_in, const int* in_sizes, int n_in,
                              void* d_out, int out_size, void* d_ws, size_t ws_size,
                              hipStream_t stream) {
  const float* query   = (const float*)d_in[0];
  const float* kcache  = (const float*)d_in[1];
  const float* vcache  = (const float*)d_in[2];
  const int*   bidx    = (const int*)d_in[3];
  const int*   seqlens = (const int*)d_in[4];
  const int*   btab    = (const int*)d_in[5];
  float*       out     = (float*)d_out;
  (void)in_sizes; (void)n_in; (void)out_size; (void)d_ws; (void)ws_size;

  sparse_flash_attn_kernel<<<dim3(BATCH * HKV), dim3(NTHREADS), 0, stream>>>(
      query, kcache, vcache, bidx, seqlens, btab, out);
}
